// SO2Convolution_84859963834919
// MI455X (gfx1250) — matrix operand order repaired and checked by reading, not compiled, not run on hardware
//
#include <hip/hip_runtime.h>

typedef __attribute__((ext_vector_type(16))) __bf16 v16bf;
typedef __attribute__((ext_vector_type(8)))  float  v8f;
typedef __attribute__((ext_vector_type(4)))  float  v4f;

#define E_EDGES 200000
#define ROW_F   928   // 29 * 32 floats per edge row (inputs and outputs share layout)

// ---------------------------------------------------------------------------
// Weight prep: build combined complex weight [[Wl, Wh], [-Wh, Wl]] (or plain W0)
// in bf16, pre-swizzled into WMMA B-fragment order:
//   tile t = kTile*(N/16) + nTile holds 512 bf16:  [lane 0..31][elem 0..15]
//   k = kTile*32 + (lane<16 ? 0 : 8) + (elem&7) + (elem>>3)*16,  n = nTile*16 + (lane & 15)
// srcRows = K for the plain m=0 block, K/2 for combined complex blocks.
// ---------------------------------------------------------------------------
__global__ void prep_weights(const float* __restrict__ W, __bf16* __restrict__ dst,
                             int K, int N, int srcRows) {
  int total = K * N;
  for (int idx = blockIdx.x * blockDim.x + threadIdx.x; idx < total;
       idx += gridDim.x * blockDim.x) {
    int t    = idx >> 9;
    int r    = idx & 511;
    int lane = r >> 4;
    int elem = r & 15;
    int ntiles = N >> 4;
    int kTile  = t / ntiles;
    int nTile  = t - kTile * ntiles;
    int k = kTile * 32 + ((lane >> 4) << 3) + (elem & 7) + ((elem >> 3) << 4);
    int n = nTile * 16 + (lane & 15);
    float v;
    if (k < srcRows) {
      v = W[k * N + n];                         // [Wl | Wh] row as-is
    } else {
      int kk   = k - srcRows;
      int half = N >> 1;
      v = (n < half) ? -W[kk * N + half + n]    // -Wh
                     :  W[kk * N + (n - half)]; //  Wl
    }
    dst[idx] = (__bf16)v;
  }
}

// ---------------------------------------------------------------------------
// Per-segment GEMM: out[e, OFF + n] = sum_k feats[e, OFF + k] * Wc[k, n] (+ bias)
// Block = 256 threads = 8 wave32, M-tile = 128 edges (16 rows per wave).
// A (fp32 -> bf16, nontemporal reads) held in registers for the whole K.
// B staged per n-tile in a double-buffered LDS block with register prefetch:
// one barrier per n-tile, global weight fetch overlapped with WMMA compute.
// ---------------------------------------------------------------------------
template <int K, int N, int OFF, bool BIAS>
__global__ __launch_bounds__(256)
void so2_gemm(const float* __restrict__ feats, const __bf16* __restrict__ wseg,
              const float* __restrict__ bias, float* __restrict__ out) {
  constexpr int KT = K / 32;                    // k-tiles of 32
  constexpr int NT = N / 16;                    // n-tiles of 16
  constexpr int CHUNKS = KT * 64;               // 16B chunks per n-tile column block
  constexpr int PT = (CHUNKS + 255) / 256;      // staging chunks per thread
  __shared__ __attribute__((aligned(64))) __bf16 bsm[2 * KT * 512];

  const int tid    = threadIdx.x;
  const int wave   = tid >> 5;
  const int lane   = tid & 31;
  const int blockM = blockIdx.x * 128;

  int e = blockM + wave * 16 + (lane & 15);
  if (e >= E_EDGES) e = E_EDGES - 1;            // clamp reads; stores handled below

  // A-fragment layout (16-bit A 16x32): lane<16 holds K {0..7, 16..23},
  // lane>=16 holds K {8..15, 24..31}; row M = lane & 15.
  const float* rowp = feats + (long)e * ROW_F + OFF + ((lane >> 4) << 3);

  v16bf a[KT];
#pragma unroll
  for (int kt = 0; kt < KT; ++kt) {
    v4f f0 = __builtin_nontemporal_load((const v4f*)(rowp + kt * 32 + 0));
    v4f f1 = __builtin_nontemporal_load((const v4f*)(rowp + kt * 32 + 4));
    v4f f2 = __builtin_nontemporal_load((const v4f*)(rowp + kt * 32 + 16));
    v4f f3 = __builtin_nontemporal_load((const v4f*)(rowp + kt * 32 + 20));
    float t[16];
    *(v4f*)(t + 0)  = f0;
    *(v4f*)(t + 4)  = f1;
    *(v4f*)(t + 8)  = f2;
    *(v4f*)(t + 12) = f3;
    v16bf av;
#pragma unroll
    for (int i = 0; i < 16; ++i) av[i] = (__bf16)t[i];
    a[kt] = av;
  }

  const int  colLane  = lane & 15;
  const int  rowAdd   = (lane >> 4) << 3;       // C/D layout: upper lanes hold M+8
  const bool fullTile = (blockM + 128 <= E_EDGES);

  uint4 stage[PT];

  // Prologue: fetch + stage n-tile 0 into LDS buffer 0.
#pragma unroll
  for (int j = 0; j < PT; ++j) {
    int i = tid + j * 256;
    if (i < CHUNKS) {
      int kt = i >> 6, c = i & 63;
      stage[j] = ((const uint4*)(wseg + (long)(kt * NT + 0) * 512))[c];
    }
  }
#pragma unroll
  for (int j = 0; j < PT; ++j) {
    int i = tid + j * 256;
    if (i < CHUNKS) {
      int kt = i >> 6, c = i & 63;
      ((uint4*)(bsm + kt * 512))[c] = stage[j];
    }
  }

  for (int nt = 0; nt < NT; ++nt) {
    __syncthreads();                            // buf[nt&1] ready for all waves

    // Prefetch next n-tile's weight block into registers (overlaps compute).
    if (nt + 1 < NT) {
#pragma unroll
      for (int j = 0; j < PT; ++j) {
        int i = tid + j * 256;
        if (i < CHUNKS) {
          int kt = i >> 6, c = i & 63;
          stage[j] = ((const uint4*)(wseg + (long)(kt * NT + (nt + 1)) * 512))[c];
        }
      }
    }

    // WMMA chain, software-pipelined B-fragment loads from LDS.
    const __bf16* bb = bsm + (nt & 1) * (KT * 512);
    v8f acc = {};
    v16bf bcur = *(const v16bf*)(bb + lane * 16);
#pragma unroll
    for (int kt = 0; kt < KT; ++kt) {
      v16bf bnxt = bcur;
      if (kt + 1 < KT) bnxt = *(const v16bf*)(bb + (kt + 1) * 512 + lane * 16);
      acc = __builtin_amdgcn_wmma_f32_16x16x32_bf16(
          /*neg_a=*/false, a[kt], /*neg_b=*/false, bcur,
          /*c_mod=*/(short)0, acc, /*reuse_a=*/false, /*reuse_b=*/false);
      bcur = bnxt;
    }

    // Stage next n-tile into the other LDS buffer (read at nt+1, after barrier).
    if (nt + 1 < NT) {
      __bf16* bw = bsm + ((nt + 1) & 1) * (KT * 512);
#pragma unroll
      for (int j = 0; j < PT; ++j) {
        int i = tid + j * 256;
        if (i < CHUNKS) {
          int kt = i >> 6, c = i & 63;
          ((uint4*)(bw + kt * 512))[c] = stage[j];
        }
      }
    }

    // C writeback (nontemporal: outputs are written exactly once, no reuse).
    float bv = 0.0f;
    if (BIAS) bv = bias[nt * 16 + colLane];
    const long col   = OFF + nt * 16 + colLane;
    const int  rbase = blockM + wave * 16 + rowAdd;
    if (fullTile) {
#pragma unroll
      for (int i = 0; i < 8; ++i) {
        __builtin_nontemporal_store(acc[i] + bv,
                                    out + (long)(rbase + i) * ROW_F + col);
      }
    } else {
#pragma unroll
      for (int i = 0; i < 8; ++i) {
        int er = rbase + i;
        if (er < E_EDGES) out[(long)er * ROW_F + col] = acc[i] + bv;
      }
    }
  }
}

extern "C" void kernel_launch(void* const* d_in, const int* in_sizes, int n_in,
                              void* d_out, int out_size, void* d_ws, size_t ws_size,
                              hipStream_t stream) {
  const float* feats = (const float*)d_in[0];
  // d_in[1] = edge_embeds: unused by the reference computation (shape only)
  const float* W0 = (const float*)d_in[2];
  const float* b0 = (const float*)d_in[3];
  const float* W1 = (const float*)d_in[4];
  const float* W2 = (const float*)d_in[5];
  float* out  = (float*)d_out;
  __bf16* ws  = (__bf16*)d_ws;

  // bf16 workspace layout (elements): seg0 @0 (224*224), seg1 @50176 (384*384),
  // seg2 @197632 (320*320); total 300032 bf16 = ~586 KB (L2-resident).
  prep_weights<<<196, 256, 0, stream>>>(W0, ws + 0,      224, 224, 224);
  prep_weights<<<576, 256, 0, stream>>>(W1, ws + 50176,  384, 384, 192);
  prep_weights<<<400, 256, 0, stream>>>(W2, ws + 197632, 320, 320, 160);

  const int blocks = (E_EDGES + 127) / 128;   // 1563
  so2_gemm<224, 224,   0, true ><<<blocks, 256, 0, stream>>>(feats, ws + 0,      b0,      out);
  so2_gemm<384, 384, 224, false><<<blocks, 256, 0, stream>>>(feats, ws + 50176,  nullptr, out);
  so2_gemm<320, 320, 608, false><<<blocks, 256, 0, stream>>>(feats, ws + 197632, nullptr, out);
}
